// GraphSAGELayer_72773925863488
// MI455X (gfx1250) — matrix operand order repaired and checked by reading, not compiled, not run on hardware
//
#include <hip/hip_runtime.h>
#include <stdint.h>

typedef __bf16 bf16;
typedef __bf16   v16bf __attribute__((ext_vector_type(16)));
typedef float    v8f   __attribute__((ext_vector_type(8)));
typedef uint32_t u32x4 __attribute__((ext_vector_type(4)));

union ABFrag { v16bf v; u32x4 q[2]; };
union Pack8  { bf16 b[8]; u32x4 q; };

#define IN_DIM 128
#define ROWS_PER_BLOCK 64
#define KTILE 32

// CDNA5 async global->LDS copy (16B per lane), tracked by ASYNCcnt.
__device__ __forceinline__ void async_copy_b128(uint32_t lds_byte_addr, const void* gaddr) {
    asm volatile("global_load_async_to_lds_b128 %0, %1, off"
                 :: "v"(lds_byte_addr), "v"(gaddr) : "memory");
}
__device__ __forceinline__ void wait_asynccnt0() {
    asm volatile("s_wait_asynccnt 0x0" ::: "memory");
}

// ---------------------------------------------------------------------------
// Pass 1: f32 -> bf16 conversions into workspace.
//   hB  : h row-major bf16         [n][128]   (A-fragments of epilogue GEMM)
//   hT  : h transposed bf16        [128][n]   (B-fragments of main GEMM, contiguous K)
//   WTs/WTn : W transposed bf16    [128(N)][128(K)]
// ---------------------------------------------------------------------------
__global__ __launch_bounds__(256) void sage_convert(
    const float* __restrict__ h, const float* __restrict__ Ws, const float* __restrict__ Wn,
    bf16* __restrict__ hB, bf16* __restrict__ hT,
    bf16* __restrict__ WTs, bf16* __restrict__ WTn, int n)
{
    int idx = blockIdx.x * blockDim.x + threadIdx.x;
    int total = n * IN_DIM;
    if (idx < total) {
        int r = idx >> 7, c = idx & 127;
        bf16 b = (bf16)h[idx];
        hB[idx] = b;
        hT[(size_t)c * n + r] = b;
    }
    if (idx < IN_DIM * IN_DIM) {
        int r = idx >> 7, c = idx & 127;
        WTs[c * IN_DIM + r] = (bf16)Ws[idx];
        WTn[c * IN_DIM + r] = (bf16)Wn[idx];
    }
}

// ---------------------------------------------------------------------------
// Pass 2: fused  out = h@Ws + ((adj@h)/deg)@Wn   for a 64-row block.
// 8 wave32s: wave = (strip 0..3) x (column-half 0..1); each wave: 16 rows x 64 cols.
// Double-buffered staging: adj tile via VALU cvt path, h slab via async-to-LDS.
// ---------------------------------------------------------------------------
__global__ __launch_bounds__(256) void sage_main(
    const float* __restrict__ adj, const bf16* __restrict__ hB, const bf16* __restrict__ hT,
    const bf16* __restrict__ WTs, const bf16* __restrict__ WTn,
    float* __restrict__ out, int n)
{
    // [ sA: 2 x 64x32 bf16 = 8K ][ sB: 2 x 128x32 bf16 = 16K ]   (24K stage)
    // [ sHN: 64x128 bf16 = 16K ]                                 (epilogue overlay)
    __shared__ __align__(16) char smem[24 * 1024];
    __shared__ float s_partial[256];
    __shared__ float s_degs[ROWS_PER_BLOCK];

    bf16* sA  = (bf16*)smem;                 // [2][64][32]
    bf16* sB  = (bf16*)(smem + 8192);        // [2][128][32] col-major slabs
    bf16* sHN = (bf16*)smem;                 // [64][128]

    const int tid   = threadIdx.x;
    const int lane  = tid & 31;
    const int wave  = tid >> 5;
    const int strip = wave >> 1;             // 16-row strip 0..3
    const int half  = wave & 1;              // column half (nt 0..3 or 4..7)
    const int row0  = blockIdx.x * ROWS_PER_BLOCK;

    const int lm  = lane & 15;
    const int hi8 = (lane & 16) ? 8 : 0;     // C/D: lanes 16..31 hold M+8
    const int kbA = (lane < 16) ? 0 : 8;     // 16-bit A layout K-base per half-wave
    const int kbB = (lane < 16) ? 0 : 8;     // 16-bit B layout K-base per half-wave

    v8f acc[4];
#pragma unroll
    for (int i = 0; i < 4; ++i) acc[i] = (v8f){0,0,0,0,0,0,0,0};

    float degp = 0.0f;

    // staging assignments
    const int rS  = tid >> 2;                // adj stage row 0..63
    const int cS  = (tid & 3) * 8;           // adj stage col chunk (8 f32)
    const int cB  = tid >> 1;                // h slab column 0..127
    const int kB2 = (tid & 1) * 16;          // h slab K-half (16 bf16 = 32B)

    const uint32_t sB_base = (uint32_t)(uintptr_t)sB;   // LDS byte address

    auto stage = [&](int k0s, int buf) {
        // adj tile: f32 -> bf16 via VALU, accumulate deg partials
        const float* ap = adj + (size_t)(row0 + rS) * n + (size_t)k0s + cS;
        float4 v0 = ((const float4*)ap)[0];
        float4 v1 = ((const float4*)ap)[1];
        degp += v0.x + v0.y + v0.z + v0.w + v1.x + v1.y + v1.z + v1.w;
        Pack8 pk;
        pk.b[0] = (bf16)v0.x; pk.b[1] = (bf16)v0.y; pk.b[2] = (bf16)v0.z; pk.b[3] = (bf16)v0.w;
        pk.b[4] = (bf16)v1.x; pk.b[5] = (bf16)v1.y; pk.b[6] = (bf16)v1.z; pk.b[7] = (bf16)v1.w;
        *(u32x4*)(sA + buf * (ROWS_PER_BLOCK * KTILE) + rS * KTILE + cS) = pk.q;

        // h slab (already bf16, K contiguous): async global -> LDS, 2 x 16B per thread
        uint32_t ldsb = sB_base + (uint32_t)(buf * (IN_DIM * KTILE) + cB * KTILE + kB2) * 2u;
        const bf16* gsrc = hT + (size_t)cB * n + (size_t)k0s + kB2;
        async_copy_b128(ldsb,       gsrc);
        async_copy_b128(ldsb + 16u, gsrc + 8);

        if (k0s + KTILE < n) __builtin_prefetch(ap + KTILE, 0, 1);
    };

    // prologue: stage tile 0 into buffer 0
    stage(0, 0);
    wait_asynccnt0();
    __syncthreads();

    const int nIter = n / KTILE;
    for (int it = 0; it < nIter; ++it) {
        const int cur = it & 1;
        if (it + 1 < nIter) stage((it + 1) * KTILE, cur ^ 1);

        // A fragment (16x32 bf16): two 16B chunks per lane
        ABFrag a;
        const bf16* arow = sA + cur * (ROWS_PER_BLOCK * KTILE) + (strip * 16 + lm) * KTILE;
        a.q[0] = *(const u32x4*)(arow + kbA);
        a.q[1] = *(const u32x4*)(arow + kbA + 16);

        const bf16* sBc = sB + cur * (IN_DIM * KTILE);
#pragma unroll
        for (int nt = 0; nt < 4; ++nt) {
            int c = (half * 4 + nt) * 16 + lm;
            const bf16* bcol = sBc + c * KTILE;
            ABFrag b;
            b.q[0] = *(const u32x4*)(bcol + kbB);
            b.q[1] = *(const u32x4*)(bcol + kbB + 16);
            acc[nt] = __builtin_amdgcn_wmma_f32_16x16x32_bf16(
                false, a.v, false, b.v, (short)0, acc[nt], false, false);
        }
        wait_asynccnt0();      // drain this thread's async stores before the barrier
        __syncthreads();
    }

    // ---- deg = max(rowsum, 1): 4 partials per row ----
    s_partial[tid] = degp;
    __syncthreads();
    if (tid < ROWS_PER_BLOCK) {
        float s = s_partial[4*tid] + s_partial[4*tid+1] + s_partial[4*tid+2] + s_partial[4*tid+3];
        s_degs[tid] = fmaxf(s, 1.0f);
    }
    __syncthreads();

    // ---- scale by 1/deg, write h_neigh bf16 tile into LDS (overwrites stage bufs) ----
    float rdeg[8];
#pragma unroll
    for (int e = 0; e < 8; ++e) rdeg[e] = 1.0f / s_degs[strip * 16 + e + hi8];
#pragma unroll
    for (int nt = 0; nt < 4; ++nt) {
        int c = (half * 4 + nt) * 16 + lm;
#pragma unroll
        for (int e = 0; e < 8; ++e)
            sHN[(strip * 16 + e + hi8) * IN_DIM + c] = (bf16)(acc[nt][e] * rdeg[e]);
    }
    __syncthreads();

    // ---- epilogue: out = h@Ws + h_neigh@Wn  (K = 128, 4 WMMA K-steps each) ----
    v8f o[4];
#pragma unroll
    for (int i = 0; i < 4; ++i) o[i] = (v8f){0,0,0,0,0,0,0,0};

    const size_t mg = (size_t)(row0 + strip * 16 + lm);
#pragma unroll
    for (int k0 = 0; k0 < IN_DIM; k0 += KTILE) {
        ABFrag aS, aN;
        const bf16* hrow = hB + mg * IN_DIM + k0;                  // h rows: global
        aS.q[0] = *(const u32x4*)(hrow + kbA);
        aS.q[1] = *(const u32x4*)(hrow + kbA + 16);
        const bf16* nrow = sHN + (strip * 16 + lm) * IN_DIM + k0;  // h_neigh: LDS
        aN.q[0] = *(const u32x4*)(nrow + kbA);
        aN.q[1] = *(const u32x4*)(nrow + kbA + 16);

#pragma unroll
        for (int nt = 0; nt < 4; ++nt) {
            int c = (half * 4 + nt) * 16 + lm;
            ABFrag bS, bN;
            const bf16* wsp = WTs + c * IN_DIM + k0;               // W^T: contiguous K
            bS.q[0] = *(const u32x4*)(wsp + kbB);
            bS.q[1] = *(const u32x4*)(wsp + kbB + 16);
            const bf16* wnp = WTn + c * IN_DIM + k0;
            bN.q[0] = *(const u32x4*)(wnp + kbB);
            bN.q[1] = *(const u32x4*)(wnp + kbB + 16);
            o[nt] = __builtin_amdgcn_wmma_f32_16x16x32_bf16(
                false, aS.v, false, bS.v, (short)0, o[nt], false, false);
            o[nt] = __builtin_amdgcn_wmma_f32_16x16x32_bf16(
                false, aN.v, false, bN.v, (short)0, o[nt], false, false);
        }
    }

    // ---- store f32 output per C-layout: VGPR e -> row e(+8), lane&15 -> col ----
#pragma unroll
    for (int nt = 0; nt < 4; ++nt) {
        int c = (half * 4 + nt) * 16 + lm;
#pragma unroll
        for (int e = 0; e < 8; ++e)
            out[(size_t)(row0 + strip * 16 + e + hi8) * IN_DIM + c] = o[nt][e];
    }
}

// ---------------------------------------------------------------------------
extern "C" void kernel_launch(void* const* d_in, const int* in_sizes, int n_in,
                              void* d_out, int out_size, void* d_ws, size_t ws_size,
                              hipStream_t stream) {
    const float* h   = (const float*)d_in[0];
    const float* adj = (const float*)d_in[1];
    const float* Ws  = (const float*)d_in[2];
    const float* Wn  = (const float*)d_in[3];
    float* out = (float*)d_out;

    const int n = in_sizes[0] / IN_DIM;            // 16384
    const size_t hElems = (size_t)n * IN_DIM;

    char* ws = (char*)d_ws;                        // ~8.06 MiB used
    bf16* hT  = (bf16*)ws;                                      // [128][n]
    bf16* hBf = (bf16*)(ws + hElems * 2);                       // [n][128]
    bf16* WTs = (bf16*)(ws + hElems * 4);                       // [128][128]
    bf16* WTn = (bf16*)(ws + hElems * 4 + IN_DIM * IN_DIM * 2); // [128][128]

    int cvtBlocks = (int)((hElems + 255) / 256);
    sage_convert<<<cvtBlocks, 256, 0, stream>>>(h, Ws, Wn, hBf, hT, WTs, WTn, n);
    sage_main<<<n / ROWS_PER_BLOCK, 256, 0, stream>>>(adj, hBf, hT, WTs, WTn, out, n);
}
